// Decoder_69776038690924
// MI455X (gfx1250) — matrix operand order repaired and checked by reading, not compiled, not run on hardware
//
#include <hip/hip_runtime.h>
#include <hip/hip_fp16.h>

// ---------- problem constants ----------
#define BATCH   32768
#define HID     128
#define TLEN    12
#define G4      512            // 4*HID
#define MROWS   (BATCH * TLEN) // 393216
#define NPAD    64             // Linear1 50 -> padded 64
#define EPSBN   1e-5f

// ---------- vector types ----------
typedef _Float16 v16h __attribute__((ext_vector_type(16)));
typedef _Float16 h8   __attribute__((ext_vector_type(8)));
typedef _Float16 h4   __attribute__((ext_vector_type(4)));
typedef float    v8f  __attribute__((ext_vector_type(8)));
typedef unsigned int u32x4 __attribute__((ext_vector_type(4)));
typedef int          i32x4 __attribute__((ext_vector_type(4)));
typedef int          i32x8 __attribute__((ext_vector_type(8)));

// ---------- workspace layout (bytes) ----------
#define WC_OFF     0u                       // f16 512x128            (131072)
#define BC_OFF     131072u                  // f32 512                (2048)
#define W1F_OFF    133120u                  // f16 64x128             (16384)
#define SCAL_OFF   149504u                  // f32 128: [6]a1 [7]d1 [8]a2 [9]d2 [10]a3 [11]d3 [12,13]cO [16..79]cY
#define PART1_OFF  150528u                  // f32 512*2
#define PART2_OFF  154624u                  // f32 6144*2
#define PART3_OFF  203776u                  // f32 3072*2
#define HS_OFF     262144u                  // f16 393216x128  (100663296) -- dead after mm1
#define Y_OFF      100925440u               // f16 393216x64   (50331648)
#define R_OFF      262144u                  // f16 393216x64 (aliases HS region)

#define NB_LSTM 512
#define NB_MM1  6144
#define NB_RELU 3072

__device__ __forceinline__ float sigm(float x) { return 1.0f / (1.0f + __expf(-x)); }
__device__ __forceinline__ float ftanh(float x) {
  float e = __expf(-2.0f * x);               // single v_exp_f32
  return (1.0f - e) / (1.0f + e);
}

// =====================================================================
// Tensor Data Mover: 2D tile (rows x 32 8-byte units = rows x 128 f16)
// global -> LDS, padding 16B after every 256B row (LDS stride 272B).
// D# per CDNA5 ISA ch.8: group0 {count|lds_addr|global_addr|type=2},
// group1 {data_size=8B, pad_enable, pad_interval=64DW, pad_amount=4DW,
//          tensor_dim0/1, tile_dim0/1, tensor_dim0_stride}.
// =====================================================================
__device__ __forceinline__ void tdm_load_2d_pad(unsigned lds_off, const void* gptr,
                                                unsigned rows) {
  const unsigned units = 32u;                // 32 x 8B = 256B per row
  size_t ga = (size_t)gptr;
  u32x4 g0;
  g0[0] = 1u;                                         // count=1, user descriptor
  g0[1] = lds_off;                                    // LDS byte address
  g0[2] = (unsigned)ga;                               // global addr [31:0]
  g0[3] = (unsigned)((ga >> 32) & 0x1FFFFFFu) | (2u << 30);  // [56:32] | type=2
  i32x8 g1;
  g1[0] = (int)((3u << 16) | (1u << 20) | (5u << 22) | (3u << 25));
  //            data_size=8B  pad_enable  pad_int=64DW  pad_amt=4DW
  g1[1] = (int)(units << 16);                         // tensor_dim0[15:0]
  g1[2] = (int)(rows << 16);                          // tensor_dim1[15:0]
  g1[3] = (int)(units << 16);                         // tile_dim0
  g1[4] = (int)rows;                                  // tile_dim1 (tile_dim2=0)
  g1[5] = (int)units;                                 // tensor_dim0_stride[31:0]
  g1[6] = 0;
  g1[7] = 0;
  i32x4 gz = {0, 0, 0, 0};
#if __clang_major__ >= 23
  i32x8 gz8 = {0, 0, 0, 0, 0, 0, 0, 0};
  __builtin_amdgcn_tensor_load_to_lds(g0, g1, gz, gz, gz8, 0);
#else
  __builtin_amdgcn_tensor_load_to_lds(g0, g1, gz, gz, 0);
#endif
}

// =====================================================================
// Kernel 0: fold weights.  Wc = W_ih + W_hh (f16), bc = b_ih + b_hh,
//           W1f = W1 zero-padded 50->64 rows (f16).
// =====================================================================
__global__ __launch_bounds__(256) void prep_kernel(
    const float* __restrict__ W_ih, const float* __restrict__ W_hh,
    const float* __restrict__ b_ih, const float* __restrict__ b_hh,
    const float* __restrict__ W1,
    _Float16* __restrict__ Wc, float* __restrict__ bc,
    _Float16* __restrict__ W1f, float* __restrict__ scal) {
  int idx = blockIdx.x * 256 + threadIdx.x;       // 65536 threads
  if (idx < G4 * HID) Wc[idx] = (_Float16)(W_ih[idx] + W_hh[idx]);
  if (idx < G4)       bc[idx] = b_ih[idx] + b_hh[idx];
  if (idx < NPAD * HID)
    W1f[idx] = (idx < 50 * HID) ? (_Float16)W1[idx] : (_Float16)0.0f;
  if (idx < 16) scal[idx] = 0.0f;
}

// =====================================================================
// Kernel 1: fused 12-step LSTM.  4 waves/block, 16 batch rows per wave.
// Wc (512x128 f16) DMA'd to LDS by the TDM (each wave brings 128 rows);
// per-wave h/z/c tiles in LDS.  GEMM: 32 n-tiles x 4 k-steps of
// v_wmma_f32_16x16x32_f16, two independent n-tile chains interleaved.
// =====================================================================
#define WSTR 136   // f16 stride (272B -> 68 dwords, bank stride 4)
#define ZSTR 517   // f32 stride (bank stride 5, coprime with 64)
#define CSTR 132   // f32 stride

#define SM_WC 0u
#define SM_BC 139264u
#define SM_H  141312u           // 4 * 16*WSTR f16  = 17408
#define SM_C  158720u           // 4 * 16*CSTR f32  = 33792
#define SM_Z  192512u           // 4 * 16*ZSTR f32  = 132352
#define LSTM_SMEM 324864u

__global__ __launch_bounds__(128) void lstm_kernel(
    const float* __restrict__ h_in, const float* __restrict__ c_in,
    const _Float16* __restrict__ Wc, const float* __restrict__ bc,
    _Float16* __restrict__ Hs, float* __restrict__ part) {
  extern __shared__ __align__(16) char smem[];
  _Float16* smWc = (_Float16*)(smem + SM_WC);
  float*    smBc = (float*)   (smem + SM_BC);
  _Float16* smH  = (_Float16*)(smem + SM_H);
  float*    smC  = (float*)   (smem + SM_C);
  float*    smZ  = (float*)   (smem + SM_Z);

  const int tid  = threadIdx.x;
  const int wave = tid >> 5, lane = tid & 31;
  const int m16  = lane & 15, g = lane >> 4;

  // --- TDM: each wave DMAs its 128-row slice of Wc into padded LDS ---
  {
    unsigned ldsWc = (unsigned)(size_t)(void*)smWc;
    tdm_load_2d_pad(ldsWc + (unsigned)wave * 128u * (WSTR * 2u),
                    Wc + wave * 128 * HID, 128u);
  }
  for (int i = tid; i < G4; i += 128) smBc[i] = bc[i];

  // --- per-wave tiles ---
  const int m0 = blockIdx.x * 64 + wave * 16;
  _Float16* myH = smH + wave * 16 * WSTR;
  float*    myC = smC + wave * 16 * CSTR;
  float*    myZ = smZ + wave * 16 * ZSTR;

  { // load h (->f16 LDS) and c (->f32 LDS); lane owns (row m16, cols g*64..)
    const int   mrow = m0 + m16;
    const float* hp = h_in + (size_t)mrow * HID + g * 64;
    const float* cp = c_in + (size_t)mrow * HID + g * 64;
#pragma unroll
    for (int j = 0; j < 64; j += 4) {
      float4 hv = *(const float4*)(hp + j);
      float4 cv = *(const float4*)(cp + j);
      float*    cd = myC + m16 * CSTR + g * 64 + j;
      cd[0] = cv.x; cd[1] = cv.y; cd[2] = cv.z; cd[3] = cv.w;
      h4 hd; hd[0] = (_Float16)hv.x; hd[1] = (_Float16)hv.y;
      hd[2] = (_Float16)hv.z; hd[3] = (_Float16)hv.w;
      *(h4*)(myH + m16 * WSTR + g * 64 + j) = hd;
    }
  }
  __builtin_amdgcn_s_wait_tensorcnt(0);      // own TDM done
  __syncthreads();                           // all waves' slices visible

  float s1 = 0.0f, q1 = 0.0f;

  // lane-fixed store base into z: constant DS-immediate offsets off this
  float* zb = myZ + (8 * g) * ZSTR + m16;

  for (int t = 0; t < TLEN; ++t) {
    // --- A fragments: 16x128 of h, 4 k-tiles of 16x32 f16 ---
    v16h a[4];
#pragma unroll
    for (int kt = 0; kt < 4; ++kt) {
      const _Float16* base = myH + m16 * WSTR + kt * 32 + g * 8;
      h8 lo = *(const h8*)base;
      h8 hi = *(const h8*)(base + 16);
#pragma unroll
      for (int i2 = 0; i2 < 8; ++i2) { a[kt][i2] = lo[i2]; a[kt][8 + i2] = hi[i2]; }
    }
    // --- z = h @ Wc^T : 32 n-tiles, two interleaved accumulator chains ---
    for (int nt = 0; nt < 32; nt += 2) {
      v8f acc0 = {}, acc1 = {};
      const _Float16* bbase = smWc + (nt * 16 + m16) * WSTR + g * 8;
#pragma unroll
      for (int kt = 0; kt < 4; ++kt) {
        const _Float16* bb0 = bbase + kt * 32;
        const _Float16* bb1 = bb0 + 16 * WSTR;
        h8 lo0 = *(const h8*)bb0, hi0 = *(const h8*)(bb0 + 16);
        h8 lo1 = *(const h8*)bb1, hi1 = *(const h8*)(bb1 + 16);
        v16h bf0, bf1;
#pragma unroll
        for (int i2 = 0; i2 < 8; ++i2) {
          bf0[i2] = lo0[i2]; bf0[8 + i2] = hi0[i2];
          bf1[i2] = lo1[i2]; bf1[8 + i2] = hi1[i2];
        }
        acc0 = __builtin_amdgcn_wmma_f32_16x16x32_f16(false, a[kt], false, bf0,
                                                      (short)0, acc0, false, false);
        acc1 = __builtin_amdgcn_wmma_f32_16x16x32_f16(false, a[kt], false, bf1,
                                                      (short)0, acc1, false, false);
      }
      float* zg = zb + nt * 16;                 // one add per group
#pragma unroll
      for (int r2 = 0; r2 < 8; ++r2) {          // C/D layout: lane=N, vgpr=M
        zg[r2 * ZSTR]      = acc0[r2];          // const imm offsets (<64KB)
        zg[r2 * ZSTR + 16] = acc1[r2];
      }
    }
    __syncthreads();
    // --- gates (lane owns row m16, cols g*64 .. g*64+63) ---
    float* zr = myZ + m16 * ZSTR;
    const int jb = g * 64;
    const size_t hrow = ((size_t)(m0 + m16) * TLEN + t) * HID;
    for (int j2 = 0; j2 < 64; j2 += 4) {
      h4 hpack;
#pragma unroll
      for (int u = 0; u < 4; ++u) {
        int j = jb + j2 + u;
        float zi = zr[j]       + smBc[j];
        float zf = zr[128 + j] + smBc[128 + j];
        float zg2 = zr[256 + j] + smBc[256 + j];
        float zo = zr[384 + j] + smBc[384 + j];
        float cold = myC[m16 * CSTR + j];
        float c2 = sigm(zf) * cold + sigm(zi) * ftanh(zg2);
        float h2 = sigm(zo) * ftanh(c2);
        myC[m16 * CSTR + j] = c2;
        hpack[u] = (_Float16)h2;
        s1 += h2; q1 += h2 * h2;
      }
      *(h4*)(myH + m16 * WSTR + jb + j2) = hpack;   // next step's A tile
      *(h4*)(&Hs[hrow + jb + j2]) = hpack;          // global H (8B store)
    }
    __syncthreads();
  }

  // --- deterministic block partial [sum, sumsq] ---
  __shared__ float red[2][4];
#pragma unroll
  for (int off = 16; off > 0; off >>= 1) {
    s1 += __shfl_down(s1, off, 32);
    q1 += __shfl_down(q1, off, 32);
  }
  if (lane == 0) { red[0][wave] = s1; red[1][wave] = q1; }
  __syncthreads();
  if (tid == 0) {
    part[blockIdx.x * 2]     = red[0][0] + red[0][1] + red[0][2] + red[0][3];
    part[blockIdx.x * 2 + 1] = red[1][0] + red[1][1] + red[1][2] + red[1][3];
  }
}

// =====================================================================
// Kernel 2: finalize BN1 -> a1,d1; fold BN1 shift + b1 into cY[64].
// =====================================================================
__global__ void fin1_kernel(const float* __restrict__ part,
                            const float* __restrict__ W1, const float* __restrict__ b1,
                            const float* __restrict__ gamma1, const float* __restrict__ beta1,
                            float* __restrict__ scal) {
  __shared__ float sh[2][64];
  int t = threadIdx.x;                       // 64 threads
  float s = 0.0f, q = 0.0f;
  for (int i = t; i < NB_LSTM; i += 64) { s += part[2 * i]; q += part[2 * i + 1]; }
  sh[0][t] = s; sh[1][t] = q; __syncthreads();
  for (int off = 32; off > 0; off >>= 1) {
    if (t < off) { sh[0][t] += sh[0][t + off]; sh[1][t] += sh[1][t + off]; }
    __syncthreads();
  }
  const float cnt  = (float)MROWS * (float)HID;
  float mean = sh[0][0] / cnt;
  float var  = sh[1][0] / cnt - mean * mean;
  float rs   = rsqrtf(var + EPSBN);
  float a1 = rs * gamma1[0];
  float d1 = beta1[0] - mean * a1;
  if (t == 0) { scal[6] = a1; scal[7] = d1; }
  float cy = 0.0f;
  if (t < 50) {
    float rsum = 0.0f;
    for (int k = 0; k < HID; ++k) rsum += W1[t * HID + k];
    cy = d1 * rsum + b1[t];
  }
  scal[16 + t] = cy;                          // n >= 50 -> 0
}

// =====================================================================
// Kernel 3: Y = a1*(H @ W1f^T) + cY   (M=393216, K=128, N=64 padded)
// TDM stages both operand tiles into padded LDS; 4 independent WMMA
// accumulator chains; per-block BN2 partials over the 50 valid columns.
// =====================================================================
__global__ __launch_bounds__(128) void mm1_kernel(
    const _Float16* __restrict__ Hs, const _Float16* __restrict__ W1f,
    const float* __restrict__ scal, _Float16* __restrict__ Y,
    float* __restrict__ part) {
  __shared__ _Float16 smW[NPAD * WSTR];
  __shared__ _Float16 smA[64 * WSTR];
  __shared__ float red[2][4];
  const int tid = threadIdx.x, wave = tid >> 5, lane = tid & 31;
  const int m16 = lane & 15, g = lane >> 4;
  const size_t m0 = (size_t)blockIdx.x * 64;

  { // each wave DMAs 16 rows of W1f and 16 rows of the Hs tile
    unsigned ldsW = (unsigned)(size_t)(void*)smW;
    unsigned ldsA = (unsigned)(size_t)(void*)smA;
    tdm_load_2d_pad(ldsW + (unsigned)wave * 16u * (WSTR * 2u),
                    W1f + wave * 16 * HID, 16u);
    tdm_load_2d_pad(ldsA + (unsigned)wave * 16u * (WSTR * 2u),
                    Hs + (m0 + wave * 16) * HID, 16u);
  }
  const float a1 = scal[6];
  __builtin_amdgcn_s_wait_tensorcnt(0);
  __syncthreads();

  v16h a[4];
  const _Float16* myA = smA + wave * 16 * WSTR;
#pragma unroll
  for (int kt = 0; kt < 4; ++kt) {
    const _Float16* b = myA + m16 * WSTR + kt * 32 + g * 8;
    h8 lo = *(const h8*)b;
    h8 hi = *(const h8*)(b + 16);
#pragma unroll
    for (int i2 = 0; i2 < 8; ++i2) { a[kt][i2] = lo[i2]; a[kt][8 + i2] = hi[i2]; }
  }
  v8f acc[4] = {{}, {}, {}, {}};
#pragma unroll
  for (int kt = 0; kt < 4; ++kt) {            // 4 independent n-tile chains
#pragma unroll
    for (int nt = 0; nt < 4; ++nt) {
      const _Float16* bb = smW + (nt * 16 + m16) * WSTR + kt * 32 + g * 8;
      h8 lo = *(const h8*)bb;
      h8 hi = *(const h8*)(bb + 16);
      v16h bf;
#pragma unroll
      for (int i2 = 0; i2 < 8; ++i2) { bf[i2] = lo[i2]; bf[8 + i2] = hi[i2]; }
      acc[nt] = __builtin_amdgcn_wmma_f32_16x16x32_f16(false, a[kt], false, bf,
                                                       (short)0, acc[nt], false, false);
    }
  }
  float s = 0.0f, q = 0.0f;
  _Float16* yb = Y + (m0 + wave * 16 + 8 * g) * NPAD + m16;
#pragma unroll
  for (int nt = 0; nt < 4; ++nt) {
    const int n = nt * 16 + m16;
    const float cy = scal[16 + n];
#pragma unroll
    for (int r2 = 0; r2 < 8; ++r2) {
      float y = a1 * acc[nt][r2] + cy;
      yb[(size_t)r2 * NPAD + nt * 16] = (_Float16)y;
      if (n < 50) { s += y; q += y * y; }
    }
  }
#pragma unroll
  for (int off = 16; off > 0; off >>= 1) {
    s += __shfl_down(s, off, 32);
    q += __shfl_down(q, off, 32);
  }
  if (lane == 0) { red[0][wave] = s; red[1][wave] = q; }
  __syncthreads();
  if (tid == 0) {
    part[blockIdx.x * 2]     = red[0][0] + red[0][1] + red[0][2] + red[0][3];
    part[blockIdx.x * 2 + 1] = red[1][0] + red[1][1] + red[1][2] + red[1][3];
  }
}

// =====================================================================
// Kernel 4/6: finalize BN2 / BN3 scalars from partials.
// =====================================================================
__global__ void fin2_kernel(const float* __restrict__ part,
                            const float* __restrict__ gamma2, const float* __restrict__ beta2,
                            float* __restrict__ scal) {
  __shared__ float sh[2][256];
  int t = threadIdx.x;
  float s = 0.0f, q = 0.0f;
  for (int i = t; i < NB_MM1; i += 256) { s += part[2 * i]; q += part[2 * i + 1]; }
  sh[0][t] = s; sh[1][t] = q; __syncthreads();
  for (int off = 128; off > 0; off >>= 1) {
    if (t < off) { sh[0][t] += sh[0][t + off]; sh[1][t] += sh[1][t + off]; }
    __syncthreads();
  }
  if (t == 0) {
    const float cnt = (float)MROWS * 50.0f;
    float mean = sh[0][0] / cnt;
    float var  = sh[1][0] / cnt - mean * mean;
    float rs   = rsqrtf(var + EPSBN);
    float a2 = rs * gamma2[0];
    scal[8] = a2; scal[9] = beta2[0] - mean * a2;
  }
}

__global__ void fin3_kernel(const float* __restrict__ part,
                            const float* __restrict__ gamma3, const float* __restrict__ beta3,
                            const float* __restrict__ W2, const float* __restrict__ b2o,
                            float* __restrict__ scal) {
  __shared__ float sh[2][256];
  int t = threadIdx.x;
  float s = 0.0f, q = 0.0f;
  for (int i = t; i < NB_RELU; i += 256) { s += part[2 * i]; q += part[2 * i + 1]; }
  sh[0][t] = s; sh[1][t] = q; __syncthreads();
  for (int off = 128; off > 0; off >>= 1) {
    if (t < off) { sh[0][t] += sh[0][t + off]; sh[1][t] += sh[1][t + off]; }
    __syncthreads();
  }
  if (t == 0) {
    const float cnt = (float)MROWS * 50.0f;
    float mean = sh[0][0] / cnt;
    float var  = sh[1][0] / cnt - mean * mean;
    float rs   = rsqrtf(var + EPSBN);
    float a3 = rs * gamma3[0];
    float d3 = beta3[0] - mean * a3;
    scal[10] = a3; scal[11] = d3;
    float r0 = 0.0f, r1 = 0.0f;
    for (int k = 0; k < 50; ++k) { r0 += W2[k]; r1 += W2[50 + k]; }
    scal[12] = d3 * r0 + b2o[0];
    scal[13] = d3 * r1 + b2o[1];
  }
}

// =====================================================================
// Kernel 5: R = relu(a2*Y + d2), masked to 50 valid cols; BN3 partials.
// =====================================================================
__global__ __launch_bounds__(256) void relu_kernel(
    const _Float16* __restrict__ Y, const float* __restrict__ scal,
    _Float16* __restrict__ R, float* __restrict__ part) {
  const float a2 = scal[8], d2 = scal[9];
  const size_t tid0 = (size_t)blockIdx.x * 256 + threadIdx.x;
  const size_t S = (size_t)NB_RELU * 256;
  float s = 0.0f, q = 0.0f;
  for (int i = 0; i < 32; ++i) {
    size_t e = tid0 + (size_t)i * S;          // covers 393216*64 exactly
    int n = (int)(e & 63);
    float r = fmaxf(a2 * (float)Y[e] + d2, 0.0f);
    if (n >= 50) r = 0.0f;
    R[e] = (_Float16)r;
    if (n < 50) { s += r; q += r * r; }
  }
  __shared__ float sh[2][256];
  sh[0][threadIdx.x] = s; sh[1][threadIdx.x] = q; __syncthreads();
  for (int off = 128; off > 0; off >>= 1) {
    if (threadIdx.x < off) {
      sh[0][threadIdx.x] += sh[0][threadIdx.x + off];
      sh[1][threadIdx.x] += sh[1][threadIdx.x + off];
    }
    __syncthreads();
  }
  if (threadIdx.x == 0) {
    part[blockIdx.x * 2] = sh[0][0];
    part[blockIdx.x * 2 + 1] = sh[1][0];
  }
}

// =====================================================================
// Kernel 7: out = a3*(R @ W2^T) + cO      (K=50, N=2 -> plain VALU)
// =====================================================================
__global__ __launch_bounds__(256) void out_kernel(
    const _Float16* __restrict__ R, const float* __restrict__ W2,
    const float* __restrict__ scal, float* __restrict__ out) {
  size_t row = (size_t)blockIdx.x * 256 + threadIdx.x;   // 393216 rows
  if (row >= (size_t)MROWS) return;
  const float a3 = scal[10];
  const _Float16* rp = R + row * NPAD;
  float acc0 = 0.0f, acc1 = 0.0f;
  for (int k = 0; k < 50; ++k) {
    float r = (float)rp[k];
    acc0 += r * W2[k];
    acc1 += r * W2[50 + k];
  }
  out[row * 2]     = a3 * acc0 + scal[12];
  out[row * 2 + 1] = a3 * acc1 + scal[13];
}

// =====================================================================
extern "C" void kernel_launch(void* const* d_in, const int* in_sizes, int n_in,
                              void* d_out, int out_size, void* d_ws, size_t ws_size,
                              hipStream_t stream) {
  (void)in_sizes; (void)n_in; (void)out_size; (void)ws_size;
  const float* h_in   = (const float*)d_in[0];
  const float* c_in   = (const float*)d_in[1];
  const float* W_ih   = (const float*)d_in[2];
  const float* W_hh   = (const float*)d_in[3];
  const float* b_ih   = (const float*)d_in[4];
  const float* b_hh   = (const float*)d_in[5];
  const float* gamma1 = (const float*)d_in[6];
  const float* beta1  = (const float*)d_in[7];
  const float* gamma2 = (const float*)d_in[8];
  const float* beta2  = (const float*)d_in[9];
  const float* gamma3 = (const float*)d_in[10];
  const float* beta3  = (const float*)d_in[11];
  const float* W1     = (const float*)d_in[12];
  const float* b1     = (const float*)d_in[13];
  const float* W2     = (const float*)d_in[14];
  const float* b2o    = (const float*)d_in[15];
  float* out = (float*)d_out;

  char* ws = (char*)d_ws;
  _Float16* Wc   = (_Float16*)(ws + WC_OFF);
  float*    bc   = (float*)   (ws + BC_OFF);
  _Float16* W1f  = (_Float16*)(ws + W1F_OFF);
  float*    scal = (float*)   (ws + SCAL_OFF);
  float*    p1   = (float*)   (ws + PART1_OFF);
  float*    p2   = (float*)   (ws + PART2_OFF);
  float*    p3   = (float*)   (ws + PART3_OFF);
  _Float16* Hs   = (_Float16*)(ws + HS_OFF);
  _Float16* Yb   = (_Float16*)(ws + Y_OFF);
  _Float16* Rb   = (_Float16*)(ws + R_OFF);

  prep_kernel<<<dim3(256), dim3(256), 0, stream>>>(W_ih, W_hh, b_ih, b_hh, W1,
                                                   Wc, bc, W1f, scal);
  lstm_kernel<<<dim3(NB_LSTM), dim3(128), LSTM_SMEM, stream>>>(h_in, c_in, Wc, bc,
                                                               Hs, p1);
  fin1_kernel<<<dim3(1), dim3(64), 0, stream>>>(p1, W1, b1, gamma1, beta1, scal);
  mm1_kernel<<<dim3(NB_MM1), dim3(128), 0, stream>>>(Hs, W1f, scal, Yb, p2);
  fin2_kernel<<<dim3(1), dim3(256), 0, stream>>>(p2, gamma2, beta2, scal);
  relu_kernel<<<dim3(NB_RELU), dim3(256), 0, stream>>>(Yb, scal, Rb, p3);
  fin3_kernel<<<dim3(1), dim3(256), 0, stream>>>(p3, gamma3, beta3, W2, b2o, scal);
  out_kernel<<<dim3(1536), dim3(256), 0, stream>>>(Rb, W2, scal, out);
}
